// SoftlyMaskedAttention_49761491091515
// MI455X (gfx1250) — matrix operand order repaired and checked by reading, not compiled, not run on hardware
//
#include <hip/hip_runtime.h>

// ---------------------------------------------------------------------------
// SoftlyMaskedAttention for MI455X (gfx1250, wave32, WMMA bf16 16x16x32)
//
//   pass 0: convert Q, K, Wq, Wk, Wv to bf16 (memory bound, ~5us)
//   q  = Qb Wqb^T + bq          [B,LQ,H]   (bf16 in ws)
//   k  = Kb Wkb^T + bk          [B,LK,H]   (bf16 in ws)
//   v  = k  Wvb^T + bv          stored transposed: vt[b][dv][lk] (bf16)
//   S = q k^T / sqrt(H) + log(mask);  P = softmax(S);  out = P v   (fp32)
//
// Attention uses a fat 64-query-row tile per workgroup (8 waves, one WGP,
// ~301KB of the 320KB LDS): K tiles double-buffered and prefetched with
// global_load_async_to_lds_b128 one iteration ahead, V tiles DMA'd during
// the score phase. This lifts arithmetic intensity 4x vs a 16-row tile
// (L2 traffic ~1GB total) so the WMMA stream, not data movement, limits.
// ---------------------------------------------------------------------------

constexpr int NB  = 8;
constexpr int NLQ = 2048;
constexpr int NLK = 2048;
constexpr int NDQ = 1024;
constexpr int NDK = 512;
constexpr int NH  = 512;
constexpr int NDV = 512;

typedef __attribute__((ext_vector_type(16))) __bf16          v16bf;
typedef __attribute__((ext_vector_type(16))) unsigned short  v16us;
typedef __attribute__((ext_vector_type(8)))  float           v8f;

union Frag {
  uint4  q[2];
  v16us  u;
  v16bf  b;
};

__device__ __forceinline__ unsigned short f2bf(float f) {
  unsigned int u = __float_as_uint(f);
  u += 0x7fffu + ((u >> 16) & 1u);          // round-to-nearest-even
  return (unsigned short)(u >> 16);
}

__device__ __forceinline__ v8f wmma_bf16(const Frag& a, const Frag& b, v8f c) {
  return __builtin_amdgcn_wmma_f32_16x16x32_bf16(
      false, a.b, false, b.b, (short)0, c, false, false);
}

// A-matrix 16x32 bf16 fragment: per lane, 16 contiguous elems at p[0..15],
// p already includes the lane's k-offset koff = (lane>=16)*16.
__device__ __forceinline__ Frag load_a_bf16(const unsigned short* p) {
  Frag f;
  f.q[0] = *(const uint4*)(p);
  f.q[1] = *(const uint4*)(p + 8);
  return f;
}

// B-matrix 32x16 bf16 fragment: per lane, 16 contiguous elems at p
// (p includes column row-offset and k-offset (lane>=16)*16).
__device__ __forceinline__ Frag load_b_bf16(const unsigned short* p) {
  Frag f;
  f.q[0] = *(const uint4*)(p);
  f.q[1] = *(const uint4*)(p + 8);
  return f;
}

// Async DMA: copy 16 bytes global -> LDS, tracked by ASYNCcnt.
__device__ __forceinline__ void async_copy_b128(unsigned lds_off,
                                                const void* gaddr) {
  asm volatile("global_load_async_to_lds_b128 %0, %1, off"
               :: "v"(lds_off), "v"(gaddr)
               : "memory");
}
__device__ __forceinline__ void wait_async0() {
  asm volatile("s_wait_asynccnt 0x0" ::: "memory");
}
__device__ __forceinline__ void wait_async16() {
  asm volatile("s_wait_asynccnt 0x10" ::: "memory");
}

// ---------------------------------------------------------------------------
// fp32 -> bf16 bulk convert: 8 elements per thread.
// ---------------------------------------------------------------------------
__global__ __launch_bounds__(256)
void cvt_bf16(const float* __restrict__ x, unsigned short* __restrict__ y,
              int n8) {
  const int i = blockIdx.x * 256 + threadIdx.x;
  if (i >= n8) return;
  const size_t e = (size_t)i * 8;
  const float4 a = *(const float4*)(x + e);
  const float4 b = *(const float4*)(x + e + 4);
  uint4 o;
  o.x = (unsigned)f2bf(a.x) | ((unsigned)f2bf(a.y) << 16);
  o.y = (unsigned)f2bf(a.z) | ((unsigned)f2bf(a.w) << 16);
  o.z = (unsigned)f2bf(b.x) | ((unsigned)f2bf(b.y) << 16);
  o.w = (unsigned)f2bf(b.z) | ((unsigned)f2bf(b.w) << 16);
  *(uint4*)(y + e) = o;
}

// ---------------------------------------------------------------------------
// Projection GEMM (all-bf16 operands):
//   Y[M,N](bf16) = X[M,KD] * W[N,KD]^T + bias[N]
// Block = 256 threads (8 waves) covers a 16(M) x 512(N) tile; wave w owns
// cols [64w, 64w+64). The 16xKD A tile is async-DMA'd to LDS once per block
// (shared by all 8 waves); W streams from L2 double-buffered in registers.
// grid.x = M/16.  OUT_T: store transposed per batch (for V): Yt[b][n][lk].
// ---------------------------------------------------------------------------
template <int KD, bool OUT_T>
__global__ __launch_bounds__(256)
void proj_gemm(const unsigned short* __restrict__ X,
               const unsigned short* __restrict__ W,
               const float* __restrict__ bias, unsigned short* __restrict__ Y,
               int M, int N, int batchLen) {
  const int lane = threadIdx.x & 31;
  const int wave = threadIdx.x >> 5;
  const int hl   = lane >> 4;
  const int l15  = lane & 15;
  const int m0   = blockIdx.x * 16;
  const int n0   = wave * 64;

  __shared__ __align__(16) unsigned short sA[16][KD + 8];

  // stage A tile via async DMA (16*KD/8 16B-chunks over 256 threads)
  {
    const unsigned short* xbase = X + (size_t)m0 * KD;
    constexpr int ROWCH = KD / 8;            // 16B chunks per row
    constexpr int PER   = 16 * ROWCH / 256;  // chunks per thread
#pragma unroll
    for (int j = 0; j < PER; ++j) {
      const int c   = threadIdx.x + j * 256;
      const int row = c / ROWCH;
      const int col = (c % ROWCH) * 8;
      async_copy_b128((unsigned)(size_t)&sA[row][col],
                      xbase + (size_t)row * KD + col);
    }
    wait_async0();
    __syncthreads();
  }

  const unsigned short* Brow = W + (size_t)(n0 + l15) * KD + hl * 16;

  v8f acc[4] = {};
  Frag bb[2][4];

  constexpr int NK = KD / 32;

#pragma unroll
  for (int j = 0; j < 4; ++j) bb[0][j] = load_b_bf16(Brow + (size_t)16 * j * KD);

#pragma unroll
  for (int t = 0; t < NK; t += 2) {
    if (t + 1 < NK) {
#pragma unroll
      for (int j = 0; j < 4; ++j)
        bb[1][j] = load_b_bf16(Brow + (size_t)16 * j * KD + (t + 1) * 32);
    }
    {
      Frag a = load_a_bf16(&sA[l15][t * 32 + hl * 16]);
#pragma unroll
      for (int j = 0; j < 4; ++j) acc[j] = wmma_bf16(a, bb[0][j], acc[j]);
    }
    if (t + 1 < NK) {
      if (t + 2 < NK) {
#pragma unroll
        for (int j = 0; j < 4; ++j)
          bb[0][j] = load_b_bf16(Brow + (size_t)16 * j * KD + (t + 2) * 32);
      }
      Frag a = load_a_bf16(&sA[l15][(t + 1) * 32 + hl * 16]);
#pragma unroll
      for (int j = 0; j < 4; ++j) acc[j] = wmma_bf16(a, bb[1][j], acc[j]);
    }
  }

#pragma unroll
  for (int j = 0; j < 4; ++j) {
    const int col = n0 + 16 * j + l15;
    const float bv = bias[col];
#pragma unroll
    for (int i = 0; i < 8; ++i) {
      const int row = m0 + i + 8 * hl;
      const unsigned short o = f2bf(acc[j][i] + bv);
      if constexpr (OUT_T) {
        const int bt = row / batchLen;
        const int lk = row - bt * batchLen;
        Y[(size_t)bt * N * batchLen + (size_t)col * batchLen + lk] = o;
      } else {
        Y[(size_t)row * N + col] = o;
      }
    }
  }
}

// ---------------------------------------------------------------------------
// Fused flash attention, fat tile. Block = 256 threads (8 waves) owns 64
// query rows; wave = (row-group rg 0..3) x (dv-half dh 0..1, 256 cols each).
// Dynamic LDS layout (300,800 B, ~one block per 320KB WGP):
//   sQ [64][520]  q tile (staged once, async DMA)
//   sK [2][64][520] K tiles, double-buffered, prefetched 1 iter ahead
//   sV [512][72]  V tile for current iteration (DMA'd during score phase)
//   sS [64][68], sP [64][72], sM/sL/sScale [64]
// Per 64-key iteration and wave: 32 score WMMAs + 32 P@V WMMAs, all
// operands via ds_load_b128. grid.x = B * LQ/64.
// ---------------------------------------------------------------------------
constexpr int QROWS   = 64;
constexpr int KSTRIDE = NH + 8;                        // 520
constexpr unsigned OFF_SQ = 0;
constexpr unsigned OFF_SK = OFF_SQ + QROWS * KSTRIDE * 2;          //  66,560
constexpr unsigned OFF_SV = OFF_SK + 2 * 64 * KSTRIDE * 2;         // 199,680
constexpr unsigned OFF_SS = OFF_SV + NDV * 72 * 2;                 // 273,408
constexpr unsigned OFF_SP = OFF_SS + QROWS * 68 * 4;               // 290,816
constexpr unsigned OFF_SM = OFF_SP + QROWS * 72 * 2;               // 300,032
constexpr unsigned OFF_SL = OFF_SM + QROWS * 4;
constexpr unsigned OFF_SC = OFF_SL + QROWS * 4;
constexpr unsigned SMEM_BYTES = OFF_SC + QROWS * 4;                // 300,800

__global__ __launch_bounds__(256)
void attn_kernel(const unsigned short* __restrict__ qb,
                 const unsigned short* __restrict__ kb,
                 const unsigned short* __restrict__ vt,
                 const float* __restrict__ mask,
                 float* __restrict__ out) {
  extern __shared__ __align__(16) unsigned char smem[];
  unsigned short (*sQ)[KSTRIDE] = (unsigned short(*)[KSTRIDE])(smem + OFF_SQ);
  unsigned short (*sK)[KSTRIDE] = (unsigned short(*)[KSTRIDE])(smem + OFF_SK);
  unsigned short (*sV)[72]      = (unsigned short(*)[72])(smem + OFF_SV);
  float          (*sS)[68]      = (float(*)[68])(smem + OFF_SS);
  unsigned short (*sP)[72]      = (unsigned short(*)[72])(smem + OFF_SP);
  float* sM     = (float*)(smem + OFF_SM);
  float* sL     = (float*)(smem + OFF_SL);
  float* sScale = (float*)(smem + OFF_SC);

  const int tid  = threadIdx.x;
  const int lane = tid & 31;
  const int wave = tid >> 5;
  const int hl   = lane >> 4;
  const int l15  = lane & 15;
  const int rg   = wave >> 1;                // row group 0..3
  const int dh   = wave & 1;                 // dv half 0..1
  const int r0   = rg * 16;
  const int dv0  = dh * 256;
  const int b    = blockIdx.x >> 5;          // LQ/64 == 32 blocks per batch
  const int q0   = (blockIdx.x & 31) * 64;

  if (tid < QROWS) { sM[tid] = -1e30f; sL[tid] = 0.0f; }

  // ---- prologue: async-stage q tile (64x512) and K tile for kb0=0 -------
  {
    const unsigned short* qbase = qb + ((size_t)b * NLQ + q0) * NH;
    const unsigned short* kbase = kb + (size_t)b * NLK * NH;
#pragma unroll
    for (int j = 0; j < 16; ++j) {           // 4096 16B chunks each
      const int c   = tid + j * 256;
      const int row = c >> 6;                // 64 chunks per row
      const int col = (c & 63) * 8;
      async_copy_b128((unsigned)(size_t)&sQ[row][col],
                      qbase + (size_t)row * NH + col);
      async_copy_b128((unsigned)(size_t)&sK[row][col],
                      kbase + (size_t)row * NH + col);
    }
    wait_async0();
    __syncthreads();
  }

  v8f oacc[16] = {};
  const float inv_sqrt_h = 0.04419417382415922f;  // 1/sqrt(512)

  for (int kb0 = 0, t = 0; kb0 < NLK; kb0 += 64, ++t) {
    const int cur = (t & 1) * 64;            // row offset of current K buffer
    const bool more = (kb0 + 64) < NLK;

    // ---- 1. async DMA: V(t) into sV, then K(t+1) into the other buffer ---
    {
      const unsigned short* vsrc = vt + (size_t)b * NDV * NLK + kb0;
#pragma unroll
      for (int j = 0; j < 16; ++j) {         // 4096 chunks: 512 rows x 128B
        const int c   = tid + j * 256;
        const int row = c >> 3;
        const int col = (c & 7) * 8;
        async_copy_b128((unsigned)(size_t)&sV[row][col],
                        vsrc + (size_t)row * NLK + col);
      }
      if (more) {
        const unsigned short* knext =
            kb + ((size_t)b * NLK + kb0 + 64) * NH;
        const int nxt = 64 - cur;
#pragma unroll
        for (int j = 0; j < 16; ++j) {
          const int c   = tid + j * 256;
          const int row = c >> 6;
          const int col = (c & 63) * 8;
          async_copy_b128((unsigned)(size_t)&sK[nxt + row][col],
                          knext + (size_t)row * NH + col);
        }
      }
    }

    // ---- 2. scores: this wave computes S[16 rows][32 keys] ---------------
    v8f s0 = {}, s1 = {};
#pragma unroll
    for (int hh = 0; hh < 16; ++hh) {
      Frag qa = load_a_bf16(&sQ[r0 + l15][hh * 32 + hl * 16]);
      Frag k0 = load_b_bf16(&sK[cur + dh * 32 + l15][hh * 32 + hl * 16]);
      s0 = wmma_bf16(qa, k0, s0);
      Frag k1 = load_b_bf16(&sK[cur + dh * 32 + 16 + l15][hh * 32 + hl * 16]);
      s1 = wmma_bf16(qa, k1, s1);
    }
    {
      const int key0 = kb0 + dh * 32 + l15;
      const float lb0 = __logf(mask[(size_t)b * NLK + key0]);
      const float lb1 = __logf(mask[(size_t)b * NLK + key0 + 16]);
#pragma unroll
      for (int i = 0; i < 8; ++i) {
        sS[r0 + i + 8 * hl][dh * 32 + l15]      = s0[i] * inv_sqrt_h + lb0;
        sS[r0 + i + 8 * hl][dh * 32 + 16 + l15] = s1[i] * inv_sqrt_h + lb1;
      }
    }
    __syncthreads();

    // ---- 3. online softmax: 4 threads per row, 16 cols each --------------
    {
      const int r  = tid >> 2;
      const int c0 = (tid & 3) * 16;
      float vv[16];
      float mx = -1e30f;
#pragma unroll
      for (int j = 0; j < 16; ++j) { vv[j] = sS[r][c0 + j]; mx = fmaxf(mx, vv[j]); }
      mx = fmaxf(mx, __shfl_xor(mx, 1, 32));
      mx = fmaxf(mx, __shfl_xor(mx, 2, 32));
      const float mold = sM[r];
      const float mnew = fmaxf(mold, mx);
      float sum = 0.0f;
#pragma unroll
      for (int j = 0; j < 16; ++j) {
        const float p = __expf(vv[j] - mnew);
        sum += p;
        sP[r][c0 + j] = f2bf(p);
      }
      sum += __shfl_xor(sum, 1, 32);
      sum += __shfl_xor(sum, 2, 32);
      if ((tid & 3) == 0) {
        const float sc = __expf(mold - mnew);
        sM[r] = mnew;
        sL[r] = sL[r] * sc + sum;
        sScale[r] = sc;
      }
    }
    // V DMA batch (oldest 16 asyncs) must be retired before the barrier so
    // every wave sees the full V tile after it; K(t+1) copies stay in flight.
    if (more) wait_async16(); else wait_async0();
    __syncthreads();

    // ---- 4. rescale running output, accumulate P @ V ---------------------
    float sc[8];
#pragma unroll
    for (int i = 0; i < 8; ++i) sc[i] = sScale[r0 + i + 8 * hl];
#pragma unroll
    for (int n = 0; n < 16; ++n)
#pragma unroll
      for (int i = 0; i < 8; ++i) oacc[n][i] *= sc[i];

#pragma unroll
    for (int kk = 0; kk < 2; ++kk) {
      Frag pa = load_a_bf16(&sP[r0 + l15][kk * 32 + hl * 16]);
#pragma unroll
      for (int n = 0; n < 16; ++n) {
        Frag vb = load_b_bf16(&sV[dv0 + 16 * n + l15][kk * 32 + hl * 16]);
        oacc[n] = wmma_bf16(pa, vb, oacc[n]);
      }
    }

    // ---- 5. publish K(t+1); order sV/sP reuse across iterations ----------
    wait_async0();
    __syncthreads();
  }

  // ---- normalize and store ---------------------------------------------
  float linv[8];
#pragma unroll
  for (int i = 0; i < 8; ++i) linv[i] = 1.0f / sL[r0 + i + 8 * hl];
#pragma unroll
  for (int n = 0; n < 16; ++n) {
    const int dv = dv0 + 16 * n + l15;
#pragma unroll
    for (int i = 0; i < 8; ++i) {
      const int row = q0 + r0 + i + 8 * hl;
      out[((size_t)b * NLQ + row) * NDV + dv] = oacc[n][i] * linv[i];
    }
  }
}

// ---------------------------------------------------------------------------
extern "C" void kernel_launch(void* const* d_in, const int* in_sizes, int n_in,
                              void* d_out, int out_size, void* d_ws, size_t ws_size,
                              hipStream_t stream) {
  (void)in_sizes; (void)n_in; (void)out_size; (void)ws_size;

  const float* Q    = (const float*)d_in[0];   // [B,LQ,DQ]
  const float* K    = (const float*)d_in[1];   // [B,LK,DK]
  const float* mask = (const float*)d_in[2];   // [B,LK]
  const float* Wq   = (const float*)d_in[3];   // [H,DQ]
  const float* bq   = (const float*)d_in[4];   // [H]
  const float* Wk   = (const float*)d_in[5];   // [H,DK]
  const float* bk   = (const float*)d_in[6];   // [H]
  const float* Wv   = (const float*)d_in[7];   // [DV,H]
  const float* bv   = (const float*)d_in[8];   // [DV]
  float* out = (float*)d_out;

  // workspace layout (bf16 elements)
  unsigned short* qb  = (unsigned short*)d_ws;                 // [B,LQ,H]
  unsigned short* kbp = qb  + (size_t)NB * NLQ * NH;           // [B,LK,H]
  unsigned short* vt  = kbp + (size_t)NB * NLK * NH;           // [B,DV,LK]
  unsigned short* Qc  = vt  + (size_t)NB * NLK * NDV;          // [B,LQ,DQ]
  unsigned short* Kc  = Qc  + (size_t)NB * NLQ * NDQ;          // [B,LK,DK]
  unsigned short* Wqc = Kc  + (size_t)NB * NLK * NDK;          // [H,DQ]
  unsigned short* Wkc = Wqc + (size_t)NH * NDQ;                // [H,DK]
  unsigned short* Wvc = Wkc + (size_t)NH * NDK;                // [DV,H]

  auto cvt = [&](const float* src, unsigned short* dst, size_t n) {
    const int n8 = (int)(n / 8);
    cvt_bf16<<<(n8 + 255) / 256, 256, 0, stream>>>(src, dst, n8);
  };
  cvt(Q,  Qc,  (size_t)NB * NLQ * NDQ);
  cvt(K,  Kc,  (size_t)NB * NLK * NDK);
  cvt(Wq, Wqc, (size_t)NH * NDQ);
  cvt(Wk, Wkc, (size_t)NH * NDK);
  cvt(Wv, Wvc, (size_t)NDV * NH);

  const int Mq = NB * NLQ;
  const int Mk = NB * NLK;

  proj_gemm<NDQ, false><<<Mq / 16, 256, 0, stream>>>(Qc, Wqc, bq, qb, Mq, NH, 0);
  proj_gemm<NDK, false><<<Mk / 16, 256, 0, stream>>>(Kc, Wkc, bk, kbp, Mk, NH, 0);
  proj_gemm<NH,  true ><<<Mk / 16, 256, 0, stream>>>(kbp, Wvc, bv, vt, Mk, NDV, NLK);

  attn_kernel<<<NB * (NLQ / 64), 256, SMEM_BYTES, stream>>>(qb, kbp, vt, mask, out);
}
